// Attention_22024592294488
// MI455X (gfx1250) — matrix operand order repaired and checked by reading, not compiled, not run on hardware
//
#include <hip/hip_runtime.h>
#include <hip/hip_bf16.h>
#include <math.h>

typedef __attribute__((ext_vector_type(16))) _Float16 v16h;
typedef __attribute__((ext_vector_type(8)))  _Float16 v8h;
typedef __attribute__((ext_vector_type(8)))  float    v8f;

constexpr int BB = 4;
constexpr int TT = 2048;
constexpr int DD = 1024;
constexpr int HH = 16;
constexpr int DHd = 64;          // DD / HH
constexpr int MROWS = BB * TT;   // 8192

__device__ __forceinline__ v8f wmma_f16(v16h a, v16h b, v8f c) {
  return __builtin_amdgcn_wmma_f32_16x16x32_f16(false, a, false, b, (short)0, c,
                                                false, false);
}

// low 32 bits of a generic pointer to a __shared__ object == LDS byte offset
__device__ __forceinline__ unsigned lds_off(const void* p) {
  return (unsigned)(size_t)p;
}

// GVS-mode async copy: 16B per lane, global -> LDS, tracked by ASYNCcnt
__device__ __forceinline__ void async_b128(unsigned lds, unsigned goff,
                                           const void* sbase) {
  asm volatile("global_load_async_to_lds_b128 %0, %1, %2"
               :: "v"(lds), "v"(goff), "s"(sbase) : "memory");
}
__device__ __forceinline__ void wait_async0() {
  asm volatile("s_wait_asynccnt 0" ::: "memory");
}

// ---------------------------------------------------------------------------
// One-shot fp32 -> f16 conversion (bandwidth bound, vectorized)
// ---------------------------------------------------------------------------
__global__ __launch_bounds__(256)
void cvt_f32_f16(const float* __restrict__ s, _Float16* __restrict__ d, int n) {
  const int i = (blockIdx.x * 256 + threadIdx.x) * 8;
  if (i >= n) return;
  const float4 a = *(const float4*)(s + i);
  const float4 b = *(const float4*)(s + i + 4);
  v8h h;
  h[0] = (_Float16)a.x; h[1] = (_Float16)a.y;
  h[2] = (_Float16)a.z; h[3] = (_Float16)a.w;
  h[4] = (_Float16)b.x; h[5] = (_Float16)b.y;
  h[6] = (_Float16)b.z; h[7] = (_Float16)b.w;
  *(v8h*)(d + i) = h;
}

// ---------------------------------------------------------------------------
// C[8192][1024] = A_f16 @ W_f16^T + bias_f32 ; OutT = f16 or f32.
// Tile 64x128, K-step 32, double-buffered async-to-LDS staging,
// one barrier per K-step. 4 waves; 8 WMMA accumulators per wave.
// ---------------------------------------------------------------------------
template <typename OutT>
__global__ __launch_bounds__(128)
void gemm_f16(const _Float16* __restrict__ A, const _Float16* __restrict__ W,
              const float* __restrict__ bias, OutT* __restrict__ C) {
  constexpr int GM = 64, GN = 128, GK = 32;
  const int bm = blockIdx.x, bn = blockIdx.y;
  const int tid  = threadIdx.x;
  const int wave = tid >> 5;
  const int lane = tid & 31;
  const int ln   = lane & 15;
  const int hl   = lane >> 4;

  __shared__ alignas(16) _Float16 sA[2][GM][GK + 8];  // row stride 80B
  __shared__ alignas(16) _Float16 sW[2][GN][GK + 8];
  constexpr unsigned ABUF = (unsigned)sizeof(_Float16) * GM * (GK + 8);
  constexpr unsigned WBUF = (unsigned)sizeof(_Float16) * GN * (GK + 8);

  v8f acc[8] = {};
  const int arow0 = bm * GM;
  const int wrow0 = bn * GN;
  const unsigned sAb = lds_off(&sA[0][0][0]);
  const unsigned sWb = lds_off(&sW[0][0][0]);
  const unsigned r4 = lane >> 2;   // 0..7 : row within 8-row slab
  const unsigned c4 = lane & 3;    // 0..3 : 16B chunk within 64B row slice

  auto stage = [&](unsigned buf, int k0) {
    // 8 slabs for A (64 rows) + 16 slabs for W (128 rows), one b128/lane each
    for (int t = wave; t < 24; t += 4) {
      if (t < 8) {
        const unsigned row = t * 8 + r4;
        async_b128(sAb + buf * ABUF + row * 80 + c4 * 16,
                   (unsigned)(arow0 + row) * (DD * 2) + k0 * 2 + c4 * 16, A);
      } else {
        const unsigned row = (t - 8) * 8 + r4;
        async_b128(sWb + buf * WBUF + row * 80 + c4 * 16,
                   (unsigned)(wrow0 + row) * (DD * 2) + k0 * 2 + c4 * 16, W);
      }
    }
  };

  stage(0, 0);
  unsigned ib = 0;
  for (int k0 = 0; k0 < DD; k0 += GK, ib ^= 1) {
    wait_async0();       // this wave's tile-i slabs have landed
    __syncthreads();     // => all waves' slabs landed; back buffer is free
    if (k0 + GK < DD) stage(ib ^ 1, k0 + GK);   // overlaps with WMMAs below

    v16h aA;
    {
      const _Float16* ar = &sA[ib][wave * 16 + ln][8 * hl];
      #pragma unroll
      for (int i = 0; i < 8; ++i) { aA[i] = ar[i]; aA[8 + i] = ar[16 + i]; }
    }
    #pragma unroll
    for (int cb = 0; cb < 8; ++cb) {
      v16h bW;
      const _Float16* wr = &sW[ib][cb * 16 + ln][8 * hl];
      #pragma unroll
      for (int i = 0; i < 16; ++i) bW[i] = wr[i < 8 ? i : i + 8];
      acc[cb] = wmma_f16(aA, bW, acc[cb]);
    }
  }

  #pragma unroll
  for (int cb = 0; cb < 8; ++cb) {
    const int col = bn * GN + cb * 16 + ln;
    const float bv = bias[col];
    #pragma unroll
    for (int r = 0; r < 8; ++r)
      C[(size_t)(bm * GM + wave * 16 + r + 8 * hl) * DD + col] =
          (OutT)(acc[cb][r] + bv);
  }
}

// ---------------------------------------------------------------------------
// Flash attention, all-f16 operands. Block: 4 waves, 64 q-rows (16/wave),
// key tile TK=64 -> 8 QK^T WMMA + 8 PV WMMA per softmax pass.
// Double-buffered async-to-LDS K/V staging; P transposed via per-wave LDS.
// ---------------------------------------------------------------------------
__global__ __launch_bounds__(128)
void flash_attn(const _Float16* __restrict__ Qp, const _Float16* __restrict__ Kp,
                const _Float16* __restrict__ Vp,
                const unsigned char* __restrict__ msk, _Float16* __restrict__ O) {
  constexpr int TQ = 64, TK = 64;
  const int qt = blockIdx.x, h = blockIdx.y, b = blockIdx.z;
  const int tid  = threadIdx.x;
  const int wave = tid >> 5;
  const int lane = tid & 31;
  const int ln   = lane & 15;
  const int hl   = lane >> 4;

  __shared__ alignas(16) _Float16 sK[2][TK][DHd + 8];   // row stride 144B
  __shared__ alignas(16) _Float16 sV[2][TK][DHd + 8];
  __shared__ alignas(16) _Float16 sP[4][16][TK + 8];
  constexpr unsigned KVBUF = (unsigned)sizeof(_Float16) * TK * (DHd + 8);

  const int q0 = qt * TQ + wave * 16;
  const unsigned bT = (unsigned)b * TT;

  // Q fragments: dh=64 -> two K=32 A-fragments, straight f16 b128 loads
  v16h aQ[2];
  {
    const _Float16* qr = Qp + ((size_t)bT + q0 + ln) * DD + h * DHd;
    #pragma unroll
    for (int f = 0; f < 2; ++f) {
      const int s = f * 32 + 8 * hl;
      #pragma unroll
      for (int i = 0; i < 8; ++i) { aQ[f][i] = qr[s + i]; aQ[f][8 + i] = qr[s + 16 + i]; }
    }
  }

  v8f oacc[4] = {};
  float mrow[8], lrow[8];
  #pragma unroll
  for (int r = 0; r < 8; ++r) { mrow[r] = -1e30f; lrow[r] = 0.f; }

  const unsigned char* mbase = msk + (size_t)b * TT * TT;
  const unsigned sKb = lds_off(&sK[0][0][0]);
  const unsigned sVb = lds_off(&sV[0][0][0]);
  const unsigned r8 = lane >> 3;   // 0..3 : row within 4-row slab
  const unsigned c8 = lane & 7;    // 0..7 : 16B chunk in 128B row

  auto stageKV = [&](unsigned buf, int kb) {
    // 16 slabs K + 16 slabs V (64 rows each, 4 rows/slab)
    for (int t = wave; t < 32; t += 4) {
      if (t < 16) {
        const unsigned row = t * 4 + r8;
        async_b128(sKb + buf * KVBUF + row * 144 + c8 * 16,
                   (bT + kb + row) * (DD * 2) + h * (DHd * 2) + c8 * 16, Kp);
      } else {
        const unsigned row = (t - 16) * 4 + r8;
        async_b128(sVb + buf * KVBUF + row * 144 + c8 * 16,
                   (bT + kb + row) * (DD * 2) + h * (DHd * 2) + c8 * 16, Vp);
      }
    }
  };

  stageKV(0, 0);
  unsigned ib = 0;
  for (int kb = 0; kb < TT; kb += TK, ib ^= 1) {
    wait_async0();
    __syncthreads();
    if (kb + TK < TT) stageKV(ib ^ 1, kb + TK);  // overlaps with compute below

    // S[j] = Q @ K^T for key columns j*16..j*16+15, j = 0..3
    v8f S[4] = {};
    #pragma unroll
    for (int f = 0; f < 2; ++f) {
      #pragma unroll
      for (int j = 0; j < 4; ++j) {
        v16h bK;
        const _Float16* kr = &sK[ib][j * 16 + ln][f * 32 + 8 * hl];
        #pragma unroll
        for (int i = 0; i < 16; ++i) bK[i] = kr[i < 8 ? i : i + 8];
        S[j] = wmma_f16(aQ[f], bK, S[j]);
      }
    }

    // scale + mask + online softmax; write P (f16) into per-wave LDS scratch
    constexpr float scale = 0.125f;  // 1/sqrt(64)
    #pragma unroll
    for (int r = 0; r < 8; ++r) {
      const int qrow = q0 + r + 8 * hl;
      const unsigned char* mp = mbase + (size_t)qrow * TT + kb;
      float sv[4];
      #pragma unroll
      for (int j = 0; j < 4; ++j) {
        sv[j] = S[j][r] * scale;
        if (mp[j * 16 + ln]) sv[j] = -1.0e6f;
      }
      float mx = fmaxf(fmaxf(sv[0], sv[1]), fmaxf(sv[2], sv[3]));
      #pragma unroll
      for (int off = 8; off; off >>= 1) mx = fmaxf(mx, __shfl_xor(mx, off, 16));
      const float mnew = fmaxf(mrow[r], mx);
      const float corr = __expf(mrow[r] - mnew);
      mrow[r] = mnew;
      float p[4], rs = 0.f;
      #pragma unroll
      for (int j = 0; j < 4; ++j) { p[j] = __expf(sv[j] - mnew); rs += p[j]; }
      #pragma unroll
      for (int off = 8; off; off >>= 1) rs += __shfl_xor(rs, off, 16);
      lrow[r] = lrow[r] * corr + rs;
      #pragma unroll
      for (int cb = 0; cb < 4; ++cb) oacc[cb][r] *= corr;
      #pragma unroll
      for (int j = 0; j < 4; ++j)
        sP[wave][r + 8 * hl][j * 16 + ln] = (_Float16)p[j];
    }
    asm volatile("s_wait_dscnt 0" ::: "memory");  // C-layout -> A-layout swap

    // P fragments: 16x64 -> two K=32 A-fragments
    v16h aP[2];
    #pragma unroll
    for (int f = 0; f < 2; ++f) {
      const _Float16* pr = &sP[wave][ln][f * 32 + 8 * hl];
      #pragma unroll
      for (int i = 0; i < 8; ++i) { aP[f][i] = pr[i]; aP[f][8 + i] = pr[16 + i]; }
    }
    // O += P @ V : dh in four 16-col chunks, K=64 in two halves
    #pragma unroll
    for (int cb = 0; cb < 4; ++cb) {
      const int d = cb * 16 + ln;
      #pragma unroll
      for (int f = 0; f < 2; ++f) {
        v16h bV;
        #pragma unroll
        for (int i = 0; i < 16; ++i) bV[i] = sV[ib][f * 32 + 8 * hl + (i < 8 ? i : i + 8)][d];
        oacc[cb] = wmma_f16(aP[f], bV, oacc[cb]);
      }
    }
  }

  #pragma unroll
  for (int r = 0; r < 8; ++r) {
    const float inv = 1.0f / lrow[r];
    const int row = q0 + r + 8 * hl;
    #pragma unroll
    for (int cb = 0; cb < 4; ++cb)
      O[((size_t)bT + row) * DD + h * DHd + cb * 16 + ln] =
          (_Float16)(oacc[cb][r] * inv);
  }
}

// ---------------------------------------------------------------------------
extern "C" void kernel_launch(void* const* d_in, const int* in_sizes, int n_in,
                              void* d_out, int out_size, void* d_ws, size_t ws_size,
                              hipStream_t stream) {
  (void)in_sizes; (void)n_in; (void)out_size; (void)ws_size;
  const float* q  = (const float*)d_in[0];
  const float* k  = (const float*)d_in[1];
  const float* v  = (const float*)d_in[2];
  const unsigned char* mask = (const unsigned char*)d_in[3];
  const float* Wq = (const float*)d_in[4];
  const float* bq = (const float*)d_in[5];
  const float* Wk = (const float*)d_in[6];
  const float* bk = (const float*)d_in[7];
  const float* Wv = (const float*)d_in[8];
  const float* bv = (const float*)d_in[9];
  const float* Wo = (const float*)d_in[10];
  const float* bo = (const float*)d_in[11];
  float* out = (float*)d_out;

  constexpr size_t NM = (size_t)MROWS * DD;  // 8M elements
  constexpr size_t NW = (size_t)DD * DD;     // 1M elements
  _Float16* p   = (_Float16*)d_ws;           // 120MB of f16 workspace
  _Float16* qh  = p;            p += NM;
  _Float16* kh  = p;            p += NM;
  _Float16* vh  = p;            p += NM;
  _Float16* wqh = p;            p += NW;
  _Float16* wkh = p;            p += NW;
  _Float16* wvh = p;            p += NW;
  _Float16* woh = p;            p += NW;
  _Float16* Qp  = p;            p += NM;
  _Float16* Kp  = p;            p += NM;
  _Float16* Vp  = p;            p += NM;
  _Float16* Ao  = p;

  const int gM = (int)(NM / 8 / 256), gW = (int)(NW / 8 / 256);
  cvt_f32_f16<<<gM, 256, 0, stream>>>(q,  qh,  (int)NM);
  cvt_f32_f16<<<gM, 256, 0, stream>>>(k,  kh,  (int)NM);
  cvt_f32_f16<<<gM, 256, 0, stream>>>(v,  vh,  (int)NM);
  cvt_f32_f16<<<gW, 256, 0, stream>>>(Wq, wqh, (int)NW);
  cvt_f32_f16<<<gW, 256, 0, stream>>>(Wk, wkh, (int)NW);
  cvt_f32_f16<<<gW, 256, 0, stream>>>(Wv, wvh, (int)NW);
  cvt_f32_f16<<<gW, 256, 0, stream>>>(Wo, woh, (int)NW);

  dim3 blk(128);
  dim3 gg(MROWS / 64, DD / 128);             // 128 x 8
  gemm_f16<_Float16><<<gg, blk, 0, stream>>>(qh, wqh, bq, Qp);
  gemm_f16<_Float16><<<gg, blk, 0, stream>>>(kh, wkh, bk, Kp);
  gemm_f16<_Float16><<<gg, blk, 0, stream>>>(vh, wvh, bv, Vp);

  dim3 fg(TT / 64, HH, BB);                  // 32 x 16 x 4
  flash_attn<<<fg, blk, 0, stream>>>(Qp, Kp, Vp, mask, Ao);

  gemm_f16<float><<<gg, blk, 0, stream>>>(Ao, woh, bo, out);
}
